// SpatialAttnBlock_42554535968887
// MI455X (gfx1250) — matrix operand order repaired and checked by reading, not compiled, not run on hardware
//
#include <hip/hip_runtime.h>

typedef __attribute__((ext_vector_type(16))) __bf16 v16bf;
typedef __attribute__((ext_vector_type(8)))  float  v8f;
typedef int v4i __attribute__((vector_size(16)));   // matches builtin param type

#define B_  4
#define NH  8
#define HD  32
#define C_  256
#define N_  2304
#define BH  (B_ * NH)
#define CCHUNK 128            // attention column chunk
#define T64 (N_ / 64)         // 36 row-supertiles per (b,head)

#if defined(__has_builtin)
#if __has_builtin(__builtin_amdgcn_global_load_async_to_lds_b128)
#define HAVE_ASYNC_LDS 1
#endif
#endif

#define AS1 __attribute__((address_space(1)))
#define AS3 __attribute__((address_space(3)))

// Stage 16 bytes global -> LDS (async DMA path on gfx1250, else plain copy)
__device__ __forceinline__ void stage16(const __bf16* g, __bf16* l) {
#ifdef HAVE_ASYNC_LDS
    __builtin_amdgcn_global_load_async_to_lds_b128(
        (AS1 v4i*)(uintptr_t)g, (AS3 v4i*)(uintptr_t)l, 0, 0);
#else
    *(uint4*)l = *(const uint4*)g;
#endif
}

__device__ __forceinline__ void stage_wait() {
#ifdef HAVE_ASYNC_LDS
#if __has_builtin(__builtin_amdgcn_s_wait_asynccnt)
    __builtin_amdgcn_s_wait_asynccnt(0);
#else
    asm volatile("s_wait_asynccnt 0" ::: "memory");
#endif
#endif
}

// Load 16 contiguous bf16 (two b128s) as a v16bf (B-tile: lane=K-row, j=N-col)
__device__ __forceinline__ v16bf ld16(const __bf16* p) {
    union { uint4 q[2]; v16bf v; } u;
    u.q[0] = *(const uint4*)(p);
    u.q[1] = *(const uint4*)(p + 8);
    return u.v;
}

// Load A-format tile row: lane (m,hi) takes K = [hi*8,hi*8+7] and [16+hi*8 ..]
// (ISA 7.12.2 16-bit A layout) -> two contiguous b128 chunks, no repacking.
__device__ __forceinline__ v16bf ld_a(const __bf16* row, int hi) {
    union { uint4 q[2]; v16bf v; } u;
    u.q[0] = *(const uint4*)(row + hi * 8);
    u.q[1] = *(const uint4*)(row + 16 + hi * 8);
    return u.v;
}

// ---------------------------------------------------------------------------
// Kernel 1: l2-normalize q,k, convert to bf16.
//   qn : [bh][n][c]   row-major   (A-matrix side of QK^T)
//   knT: [bh][c][n]   chan-major  (B-matrix side: contiguous N per channel)
//   vn : [bh][n][c]   row-major   (B-matrix side of PV: contiguous c per row)
// ---------------------------------------------------------------------------
__global__ __launch_bounds__(256) void norm_qkv(const float* __restrict__ qkv,
                                                __bf16* __restrict__ qn,
                                                __bf16* __restrict__ knT,
                                                __bf16* __restrict__ vn) {
    int r = blockIdx.x * blockDim.x + threadIdx.x;
    if (r >= BH * N_) return;
    int bh = r / N_, n = r - bh * N_;
    int b = bh >> 3, h = bh & 7;
    const float* qbase = qkv + ((size_t)b * 3 * C_ + h * HD) * N_ + n;
    const float* kbase = qbase + (size_t)C_ * N_;
    const float* vbase = qbase + (size_t)(2 * C_) * N_;

    float qv[HD], kv[HD];
    float sq = 0.f, sk = 0.f;
#pragma unroll
    for (int c = 0; c < HD; ++c) {
        float q = qbase[(size_t)c * N_];
        float k = kbase[(size_t)c * N_];
        qv[c] = q; kv[c] = k;
        sq += q * q; sk += k * k;
    }
    float qi = 1.f / fmaxf(sqrtf(sq), 1e-12f);
    float ki = 1.f / fmaxf(sqrtf(sk), 1e-12f);
    size_t o = (size_t)r * HD;
#pragma unroll
    for (int c = 0; c < HD; ++c) {
        qn[o + c] = (__bf16)(qv[c] * qi);
        knT[((size_t)bh * HD + c) * N_ + n] = (__bf16)(kv[c] * ki);
        vn[o + c] = (__bf16)vbase[(size_t)c * N_];
    }
}

// ---------------------------------------------------------------------------
// Kernel 2: proj_w f32 -> bf16
// ---------------------------------------------------------------------------
__global__ __launch_bounds__(256) void conv_w(const float* __restrict__ w,
                                              __bf16* __restrict__ wb) {
    int i = blockIdx.x * blockDim.x + threadIdx.x;
    if (i < C_ * C_) wb[i] = (__bf16)w[i];
}

// ---------------------------------------------------------------------------
// Kernel 3: flash-attention per (b,head). 4 waves/block; each wave: 16 rows.
// Output written channel-major: obT[b][h*32+c][n]  (feeds proj B-tiles wide).
// ---------------------------------------------------------------------------
__global__ __launch_bounds__(128) void attn_kernel(const __bf16* __restrict__ qn,
                                                   const __bf16* __restrict__ knT,
                                                   const __bf16* __restrict__ vn,
                                                   const float* __restrict__ temp,
                                                   __bf16* __restrict__ obT) {
    __shared__ __bf16 lds_kt[HD][CCHUNK];            // 8 KB, chan-major K tile
    __shared__ __bf16 lds_v[CCHUNK * HD];            // 8 KB, row-major V tile
    __shared__ __bf16 lds_p[4][16 * CCHUNK];         // 16 KB, per-wave P staging

    int bh   = blockIdx.x / T64;
    int t64  = blockIdx.x % T64;
    int wave = threadIdx.x >> 5;
    int lane = threadIdx.x & 31;
    int row0 = t64 * 64 + wave * 16;
    int m = lane & 15, hi = lane >> 4;
    float tscale = temp[bh & 7];

    // Q A-tile (16x32 bf16): two b128 global loads per lane
    v16bf aq = ld_a(qn + ((size_t)bh * N_ + row0 + m) * HD, hi);

    float mrow[8], lrow[8];
    v8f oacc0 = {}, oacc1 = {};
#pragma unroll
    for (int e = 0; e < 8; ++e) { mrow[e] = 0.f; lrow[e] = 0.f; }

    const __bf16* kTb = knT + (size_t)bh * HD * N_;   // [c][n]
    const __bf16* vb  = vn + (size_t)bh * N_ * HD;    // [n][c]

    int kchan = threadIdx.x >> 2, kq = threadIdx.x & 3;  // K staging mapping

    for (int nc = 0; nc < N_; nc += CCHUNK) {
        // --- cooperative staging (b128-wide, async on gfx1250) ---
        {
            // K: thread -> (channel, 32-col quarter), contiguous 64B each
            const __bf16* ks = kTb + (size_t)kchan * N_ + nc + kq * 32;
            __bf16* kd = &lds_kt[kchan][kq * 32];
            stage16(ks, kd); stage16(ks + 8, kd + 8);
            stage16(ks + 16, kd + 16); stage16(ks + 24, kd + 24);
            // V: thread -> one 32-chan row, contiguous 64B
            const __bf16* vs = vb + (size_t)(nc + threadIdx.x) * HD;
            __bf16* vd = &lds_v[threadIdx.x * HD];
            stage16(vs, vd); stage16(vs + 8, vd + 8);
            stage16(vs + 16, vd + 16); stage16(vs + 24, vd + 24);
#ifndef HAVE_ASYNC_LDS
            if (nc + CCHUNK < N_) {   // global_prefetch_b8 of next tiles
                __builtin_prefetch(ks + CCHUNK, 0, 0);
                __builtin_prefetch(vs + (size_t)CCHUNK * HD, 0, 0);
            }
#endif
        }
        stage_wait();
        __syncthreads();

        // --- S = Q * K^T : 8 WMMAs covering 16x128 ---
        v8f s[8];
#pragma unroll
        for (int t = 0; t < 8; ++t) {
            v16bf bk;   // lane%16 = N-col, elems = K (channel) in A order
            for (int kk = 0; kk < 16; ++kk) bk[kk] = lds_kt[(kk & 8) * 2 + hi * 8 + (kk & 7)][t * 16 + m];
            v8f z = {};
            s[t] = __builtin_amdgcn_wmma_f32_16x16x32_bf16(false, aq, false, bk,
                                                           (short)0, z, false, false);
        }

        // --- relu * temp, online softmax (rows = VGPR x 16-lane half) ---
        float rmax[8];
#pragma unroll
        for (int e = 0; e < 8; ++e) rmax[e] = 0.f;
#pragma unroll
        for (int t = 0; t < 8; ++t)
#pragma unroll
            for (int e = 0; e < 8; ++e) {
                float x = fmaxf(s[t][e] * tscale, 0.f);
                s[t][e] = x;
                rmax[e] = fmaxf(rmax[e], x);
            }
#pragma unroll
        for (int e = 0; e < 8; ++e)
#pragma unroll
            for (int msk = 1; msk < 16; msk <<= 1)
                rmax[e] = fmaxf(rmax[e], __shfl_xor(rmax[e], msk, 32));

        float scale[8], rsum[8];
#pragma unroll
        for (int e = 0; e < 8; ++e) {
            float mn = fmaxf(mrow[e], rmax[e]);
            scale[e] = __expf(mrow[e] - mn);
            mrow[e] = mn;
            rsum[e] = 0.f;
        }
#pragma unroll
        for (int t = 0; t < 8; ++t)
#pragma unroll
            for (int e = 0; e < 8; ++e) {
                float p = __expf(s[t][e] - mrow[e]);
                s[t][e] = p;
                rsum[e] += p;
            }
#pragma unroll
        for (int e = 0; e < 8; ++e) {
#pragma unroll
            for (int msk = 1; msk < 16; msk <<= 1)
                rsum[e] += __shfl_xor(rsum[e], msk, 32);
            lrow[e] = lrow[e] * scale[e] + rsum[e];
            oacc0[e] *= scale[e];
            oacc1[e] *= scale[e];
        }

        // --- stage P (D-format) -> row-major 16x128 bf16 (layout transpose) ---
        __bf16* pst = &lds_p[wave][0];
#pragma unroll
        for (int t = 0; t < 8; ++t)
#pragma unroll
            for (int e = 0; e < 8; ++e)
                pst[(e + hi * 8) * CCHUNK + t * 16 + m] = (__bf16)s[t][e];

        // --- O += P * V : 8 WMMAs, P tile via b128 loads, V tiles gathered per element ---
#pragma unroll
        for (int kc = 0; kc < 4; ++kc) {
            v16bf ap = ld_a(pst + m * CCHUNK + kc * 32, hi);
            v16bf bv0, bv1;
            for (int kk = 0; kk < 16; ++kk) {
                int kr = kc * 32 + (kk & 8) * 2 + hi * 8 + (kk & 7);
                bv0[kk] = lds_v[kr * HD + m];
                bv1[kk] = lds_v[kr * HD + 16 + m];
            }
            oacc0 = __builtin_amdgcn_wmma_f32_16x16x32_bf16(false, ap, false, bv0,
                                                            (short)0, oacc0, false, false);
            oacc1 = __builtin_amdgcn_wmma_f32_16x16x32_bf16(false, ap, false, bv1,
                                                            (short)0, oacc1, false, false);
        }
        __syncthreads();
    }

    // --- finalize 1/l, store channel-major obT[b][h*32+c][n] ---
    int b = bh >> 3, h = bh & 7;
#pragma unroll
    for (int e = 0; e < 8; ++e) {
        float inv = 1.f / lrow[e];
        int grow = row0 + e + hi * 8;
        size_t c0 = ((size_t)b * C_ + h * HD + m) * N_ + grow;
        obT[c0]              = (__bf16)(oacc0[e] * inv);
        obT[c0 + 16 * N_]    = (__bf16)(oacc1[e] * inv);
    }
}

// ---------------------------------------------------------------------------
// Kernel 4: 1x1 conv = 256x256 channel matmul over B*N pixels (WMMA, K=256).
// obT is channel-major so B-tiles are per-element loads, adjacent n across lanes.
// ---------------------------------------------------------------------------
__global__ __launch_bounds__(128) void proj_kernel(const __bf16* __restrict__ obT,
                                                   const __bf16* __restrict__ pw,
                                                   const float* __restrict__ pb,
                                                   float* __restrict__ out) {
    int b     = blockIdx.x / (N_ / 16);
    int ntile = blockIdx.x % (N_ / 16);
    int wave  = threadIdx.x >> 5;
    int lane  = threadIdx.x & 31;
    int m = lane & 15, hi = lane >> 4;
    int n0 = ntile * 16;

    v8f acc[4];
    v8f z = {};
#pragma unroll
    for (int i = 0; i < 4; ++i) acc[i] = z;

#pragma unroll
    for (int kc = 0; kc < 8; ++kc) {
        // B tile: lane%16 = n-column, elems = channel K in A order
        v16bf bx;
        for (int kk = 0; kk < 16; ++kk)
            bx[kk] = obT[((size_t)b * C_ + kc * 32 + (kk & 8) * 2 + hi * 8 + (kk & 7)) * N_ + n0 + m];
#pragma unroll
        for (int i = 0; i < 4; ++i) {
            int ot = wave * 4 + i;
            v16bf aw = ld_a(pw + (size_t)(ot * 16 + m) * C_ + kc * 32, hi);
            acc[i] = __builtin_amdgcn_wmma_f32_16x16x32_bf16(false, aw, false, bx,
                                                             (short)0, acc[i], false, false);
        }
    }
#pragma unroll
    for (int i = 0; i < 4; ++i) {
        int ot = wave * 4 + i;
#pragma unroll
        for (int e = 0; e < 8; ++e) {
            int o = ot * 16 + e + hi * 8;
            out[((size_t)b * C_ + o) * N_ + n0 + m] = acc[i][e] + pb[o];
        }
    }
}

// ---------------------------------------------------------------------------
extern "C" void kernel_launch(void* const* d_in, const int* in_sizes, int n_in,
                              void* d_out, int out_size, void* d_ws, size_t ws_size,
                              hipStream_t stream) {
    const float* qkv  = (const float*)d_in[0];
    const float* temp = (const float*)d_in[1];
    const float* pw   = (const float*)d_in[2];
    const float* pb   = (const float*)d_in[3];
    float* out = (float*)d_out;

    size_t per = (size_t)BH * N_ * HD;   // 2,359,296 elems per bf16 plane
    __bf16* qn  = (__bf16*)d_ws;
    __bf16* knT = qn + per;
    __bf16* vn  = knT + per;
    __bf16* obT = vn + per;              // channel-major [b][256][N]
    __bf16* pwb = obT + per;             // 65,536 elems

    norm_qkv<<<(BH * N_ + 255) / 256, 256, 0, stream>>>(qkv, qn, knT, vn);
    conv_w<<<(C_ * C_ + 255) / 256, 256, 0, stream>>>(pw, pwb);
    attn_kernel<<<BH * T64, 128, 0, stream>>>(qn, knT, vn, temp, obT);
    proj_kernel<<<B_ * (N_ / 16), 128, 0, stream>>>(obT, pwb, pb, out);
}
